// SwinTransformerV2Block_69423851372864
// MI455X (gfx1250) — matrix operand order repaired and checked by reading, not compiled, not run on hardware
//
#include <hip/hip_runtime.h>
#include <hip/hip_bf16.h>
#include <math.h>

typedef __attribute__((ext_vector_type(16))) _Float16 v16h;
typedef __attribute__((ext_vector_type(8)))  float    v8f;

#define DIM 96
#define HEADS 3
#define HD 32
#define WS 4
#define SS 2
#define RES 48
#define NTOK 64
#define NWIN_PER_B 1728
#define NWIN 3456
#define LOG100 4.6051702f

// ---------------- WMMA fragment helpers (wave32, 16x16x32 f16) ----------------
// A (16x32): lane<16 -> row m0+lane, K base 0; lane>=16 -> row m0+lane-16, K base 16.
// element e -> K=kb+e
__device__ __forceinline__ v16h frag_a_rm(const _Float16* src, int ld, int m0, int k0) {
  int lane = threadIdx.x & 31;
  const _Float16* p = src + (m0 + (lane & 15)) * ld + k0 + ((lane >> 4) << 4);
  v16h a;
#pragma unroll
  for (int e = 0; e < 16; ++e) a[e] = p[e];
  return a;
}
// B (32x16) where B[k][n] = M[n][k] (M row-major [N][K]) -> contiguous in k per lane
__device__ __forceinline__ v16h frag_b_t(const _Float16* src, int ld, int k0, int n0) {
  int lane = threadIdx.x & 31;
  const _Float16* p = src + (n0 + (lane & 15)) * ld + k0 + ((lane >> 4) << 4);
  v16h b;
#pragma unroll
  for (int e = 0; e < 16; ++e) b[e] = p[e];
  return b;
}
// D/C (16x16 f32): VGPR r -> row m0 + 8*(lane>=16) + r, col n0 + lane%16
__device__ __forceinline__ void store_tile_h(_Float16* dst, int ld, int m0, int n0, v8f d) {
  int lane = threadIdx.x & 31;
  int n = n0 + (lane & 15);
  int mb = m0 + ((lane >> 4) << 3);
#pragma unroll
  for (int r = 0; r < 8; ++r) dst[(mb + r) * ld + n] = (_Float16)d[r];
}
__device__ __forceinline__ v8f wmma16(v16h a, v16h b, v8f c) {
  return __builtin_amdgcn_wmma_f32_16x16x32_f16(false, a, false, b, (short)0, c, false, false);
}

// ---------------- weight fp32 -> f16 staging + fused qkv bias ----------------
__global__ __launch_bounds__(256) void convert_w_kernel(
    const float* __restrict__ qkvw, const float* __restrict__ projw,
    const float* __restrict__ fc1w, const float* __restrict__ fc2w,
    const float* __restrict__ q_bias, const float* __restrict__ v_bias,
    _Float16* __restrict__ dst, float* __restrict__ qkvb) {
  int i = blockIdx.x * 256 + threadIdx.x;
  if (i < 27648)        dst[i] = (_Float16)qkvw[i];
  else if (i < 36864)   dst[i] = (_Float16)projw[i - 27648];
  else if (i < 73728)   dst[i] = (_Float16)fc1w[i - 36864];
  else if (i < 110592)  dst[i] = (_Float16)fc2w[i - 73728];
  if (blockIdx.x == 0) {
    for (int n = threadIdx.x; n < 3 * DIM; n += 256)
      qkvb[n] = (n < DIM) ? q_bias[n] : ((n < 2 * DIM) ? 0.f : v_bias[n - 2 * DIM]);
  }
}

// ---------------- continuous relative position bias ----------------
__device__ __forceinline__ float cpb_coord(int a) {
  float c = (float)(a - (WS - 1)) * (8.0f / (float)(WS - 1));
  float s = (c > 0.f) ? 1.f : ((c < 0.f) ? -1.f : 0.f);
  return s * log2f(fabsf(c) + 1.f) * (1.0f / 3.0f);  // /log2(8)
}
__global__ __launch_bounds__(256) void rpb_kernel(
    const float* __restrict__ w1, const float* __restrict__ b1,
    const float* __restrict__ w2, float* __restrict__ rpb) {
  __shared__ float tbl[343 * HEADS];
  for (int u = threadIdx.x; u < 343 * HEADS; u += 256) {
    int i = u / HEADS, h = u % HEADS;
    float cv0 = cpb_coord(i / 49), cv1 = cpb_coord((i / 7) % 7), cv2 = cpb_coord(i % 7);
    float acc = 0.f;
    for (int j = 0; j < 512; ++j) {
      float hid = w1[j * 3 + 0] * cv0 + w1[j * 3 + 1] * cv1 + w1[j * 3 + 2] * cv2 + b1[j];
      hid = fmaxf(hid, 0.f);
      acc += hid * w2[h * 512 + j];
    }
    tbl[u] = acc;
  }
  __syncthreads();
  for (int u = threadIdx.x; u < HEADS * NTOK * NTOK; u += 256) {
    int h = u / (NTOK * NTOK);
    int r = u % (NTOK * NTOK);
    int i = r / NTOK, j = r % NTOK;
    int dh = (i >> 4) - (j >> 4) + 3;
    int dw = ((i >> 2) & 3) - ((j >> 2) & 3) + 3;
    int dd = (i & 3) - (j & 3) + 3;
    float v = tbl[(dh * 49 + dw * 7 + dd) * HEADS + h];
    rpb[u] = 16.f / (1.f + __expf(-v));
  }
}

// ---------------- fused window attention (shift + qkv + cos-attn + proj + LN residual) ----
__global__ __launch_bounds__(256) void swin_attn_kernel(
    const float* __restrict__ x, const _Float16* __restrict__ w16,
    const float* __restrict__ qkvb, const float* __restrict__ logit_scale,
    const float* __restrict__ proj_b, const float* __restrict__ n1g,
    const float* __restrict__ n1b, const float* __restrict__ rpb,
    float* __restrict__ out) {
  const _Float16* qkvw = w16;            // 288 x 96 row-major
  const _Float16* projw = w16 + 27648;   // 96 x 96 row-major
  constexpr int XS = 104, QS = 200, VTS = 72, AS = 68, AHS = 72, OS = 104, PS = 100;
  __shared__ _Float16 sX[NTOK * XS];          // tokens x channels (f16)
  __shared__ _Float16 sQKV[NTOK * QS];        // tokens x (q|k) (f16)
  __shared__ _Float16 sVT[DIM * VTS];         // V transposed: dim x token (f16)
  __shared__ float    sAttn[NTOK * AS];
  __shared__ _Float16 sAttnH[NTOK * AHS];
  __shared__ _Float16 sO[NTOK * OS];
  __shared__ float    sPrj[NTOK * PS];
  __shared__ int   sRow[NTOK];
  __shared__ int   sReg[NTOK];
  __shared__ float sRq[HEADS * NTOK];
  __shared__ float sRk[HEADS * NTOK];
  __shared__ float sScale[HEADS];
  __shared__ float sMu[NTOK];
  __shared__ float sRstd[NTOK];

  const int wid = blockIdx.x;
  const int b = wid / NWIN_PER_B;
  const int wr = wid % NWIN_PER_B;
  const int wh = wr / 144, ww = (wr / 12) % 12, wd = wr % 12;
  const int tid = threadIdx.x;
  const int wave = tid >> 5;

  if (tid < NTOK) {
    int th = tid >> 4, tw = (tid >> 2) & 3, td = tid & 3;
    int hs = wh * WS + th, ws = ww * WS + tw, ds = wd * WS + td;
    int h = (hs + SS) % RES, w = (ws + SS) % RES, d = (ds + SS) % RES;
    sRow[tid] = (((b * RES + h) * RES + w) * RES + d) * DIM;
    int ra = hs < RES - WS ? 0 : (hs < RES - SS ? 1 : 2);
    int rb = ws < RES - WS ? 0 : (ws < RES - SS ? 1 : 2);
    int rc = ds < RES - WS ? 0 : (ds < RES - SS ? 1 : 2);
    sReg[tid] = ra * 9 + rb * 3 + rc;
  }
  if (tid < HEADS) sScale[tid] = __expf(fminf(logit_scale[tid], LOG100));
  __syncthreads();

  // load rolled window -> f16
  for (int i = tid; i < NTOK * DIM; i += 256) {
    int t = i / DIM, c = i % DIM;
    sX[t * XS + c] = (_Float16)x[sRow[t] + c];
  }
  __syncthreads();

  // QKV: (64x96)@(96x288), 72 tiles (4m x 18n), 9 per wave, K=3 steps
  // Q|K tiles land in sQKV (row-major), V tiles land transposed in sVT.
  for (int ti = wave * 9; ti < wave * 9 + 9; ++ti) {
    int m0 = (ti & 3) << 4;
    int n0 = (ti >> 2) << 4;
    v8f acc = {};
#pragma unroll
    for (int ks = 0; ks < 3; ++ks)
      acc = wmma16(frag_a_rm(sX, XS, m0, ks * 32), frag_b_t(qkvw, DIM, ks * 32, n0), acc);
    int lane = tid & 31;
    int n = n0 + (lane & 15);
    float bn = qkvb[n];
    int mb = m0 + ((lane >> 4) << 3);
    if (n0 < 2 * DIM) {
#pragma unroll
      for (int r = 0; r < 8; ++r) sQKV[(mb + r) * QS + n] = (_Float16)(acc[r] + bn);
    } else {
#pragma unroll
      for (int r = 0; r < 8; ++r) sVT[(n - 2 * DIM) * VTS + mb + r] = (_Float16)(acc[r] + bn);
    }
  }
  __syncthreads();

  // reciprocal row norms for q and k (cosine attention)
  for (int u = tid; u < 2 * HEADS * NTOK; u += 256) {
    int isK = u >= HEADS * NTOK;
    int v = isK ? u - HEADS * NTOK : u;
    int hh = v / NTOK, t = v % NTOK;
    int base = t * QS + hh * HD + (isK ? DIM : 0);
    float s = 0.f;
#pragma unroll
    for (int c = 0; c < HD; ++c) { float q = (float)sQKV[base + c]; s += q * q; }
    float r = 1.f / fmaxf(sqrtf(s), 1e-12f);
    if (isK) sRk[v] = r; else sRq[v] = r;
  }
  __syncthreads();

  for (int hh = 0; hh < HEADS; ++hh) {
    float scale = sScale[hh];
    // QK^T: 16 tiles (4x4), 2 per wave, K=32 (1 step)
    for (int ti = wave * 2; ti < wave * 2 + 2; ++ti) {
      int m0 = (ti >> 2) << 4;
      int n0 = (ti & 3) << 4;
      v8f acc = {};
      acc = wmma16(frag_a_rm(sQKV, QS, m0, hh * HD),
                   frag_b_t(sQKV + DIM + hh * HD, QS, 0, n0), acc);
      int lane = tid & 31;
      int n = n0 + (lane & 15);
      int mb = m0 + ((lane >> 4) << 3);
      float rk = sRk[hh * NTOK + n];
#pragma unroll
      for (int r = 0; r < 8; ++r) {
        int m = mb + r;
        float v = acc[r] * sRq[hh * NTOK + m] * rk * scale;
        v += rpb[hh * NTOK * NTOK + m * NTOK + n];
        v += (sReg[m] == sReg[n]) ? 0.f : -100.f;
        sAttn[m * AS + n] = v;
      }
    }
    __syncthreads();
    // softmax: 4 lanes per row (rows align to lane quads), shuffle reductions
    {
      int row = tid >> 2, seg = (tid & 3) << 4;
      float mx = -1e30f;
#pragma unroll
      for (int j = 0; j < 16; ++j) mx = fmaxf(mx, sAttn[row * AS + seg + j]);
      mx = fmaxf(mx, __shfl_xor(mx, 1));
      mx = fmaxf(mx, __shfl_xor(mx, 2));
      float sum = 0.f;
#pragma unroll
      for (int j = 0; j < 16; ++j) {
        float e = __expf(sAttn[row * AS + seg + j] - mx);
        sAttn[row * AS + seg + j] = e;
        sum += e;
      }
      sum += __shfl_xor(sum, 1);
      sum += __shfl_xor(sum, 2);
      float rs = 1.f / sum;
#pragma unroll
      for (int j = 0; j < 16; ++j)
        sAttnH[row * AHS + seg + j] = (_Float16)(sAttn[row * AS + seg + j] * rs);
    }
    __syncthreads();
    // AV: (64x64)@(64x32), 8 tiles (4m x 2n), 1 per wave, K=2 steps; B from V^T (contiguous)
    {
      int m0 = (wave >> 1) << 4;
      int n0 = (wave & 1) << 4;
      v8f acc = {};
#pragma unroll
      for (int ks = 0; ks < 2; ++ks)
        acc = wmma16(frag_a_rm(sAttnH, AHS, m0, ks * 32),
                     frag_b_t(sVT + hh * HD * VTS, VTS, ks * 32, n0), acc);
      store_tile_h(sO, OS, m0, hh * HD + n0, acc);
    }
    __syncthreads();
  }

  // proj: (64x96)@(96x96), 24 tiles (4m x 6n), 3 per wave
  for (int ti = wave * 3; ti < wave * 3 + 3; ++ti) {
    int m0 = (ti % 4) << 4;
    int n0 = (ti / 4) << 4;
    v8f acc = {};
#pragma unroll
    for (int ks = 0; ks < 3; ++ks)
      acc = wmma16(frag_a_rm(sO, OS, m0, ks * 32), frag_b_t(projw, DIM, ks * 32, n0), acc);
    int lane = tid & 31;
    int n = n0 + (lane & 15);
    int mb = m0 + ((lane >> 4) << 3);
    float bn = proj_b[n];
#pragma unroll
    for (int r = 0; r < 8; ++r) sPrj[(mb + r) * PS + n] = acc[r] + bn;
  }
  __syncthreads();

  // LN stats + fused residual write at un-shifted coordinates
  if (tid < NTOK) {
    float s = 0.f, s2 = 0.f;
    for (int c = 0; c < DIM; ++c) { float v = sPrj[tid * PS + c]; s += v; s2 += v * v; }
    float mu = s / DIM;
    sMu[tid] = mu;
    sRstd[tid] = rsqrtf(s2 / DIM - mu * mu + 1e-5f);
  }
  __syncthreads();
  for (int i = tid; i < NTOK * DIM; i += 256) {
    int t = i / DIM, c = i % DIM;
    float v = (sPrj[t * PS + c] - sMu[t]) * sRstd[t] * n1g[c] + n1b[c];
    out[sRow[t] + c] = x[sRow[t] + c] + v;
  }
}

// ---------------- fused MLP (fc1 + GELU + fc2 + LN residual), in-place on out ----------
__global__ __launch_bounds__(256) void swin_mlp_kernel(
    const _Float16* __restrict__ w16,
    const float* __restrict__ fc1_b, const float* __restrict__ fc2_b,
    const float* __restrict__ n2g, const float* __restrict__ n2b,
    float* __restrict__ out) {
  const _Float16* w1 = w16 + 36864;  // 384 x 96 row-major
  const _Float16* w2 = w16 + 73728;  // 96 x 384 row-major
  constexpr int XS = 104, HS = 392, OS2 = 100;
  __shared__ _Float16 sX[NTOK * XS];
  __shared__ _Float16 sH[NTOK * HS];
  __shared__ float sO2[NTOK * OS2];
  __shared__ float sMu[NTOK], sRstd[NTOK];
  const int row0 = blockIdx.x * NTOK;
  const int tid = threadIdx.x, wave = tid >> 5;

  for (int i = tid; i < NTOK * DIM; i += 256) {
    int t = i / DIM, c = i % DIM;
    sX[t * XS + c] = (_Float16)out[(row0 + t) * DIM + c];
  }
  __syncthreads();

  // fc1 + exact GELU: (64x96)@(96x384), 96 tiles (4m x 24n), 12 per wave
  for (int ti = wave * 12; ti < wave * 12 + 12; ++ti) {
    int m0 = (ti & 3) << 4;
    int n0 = (ti >> 2) << 4;
    v8f acc = {};
#pragma unroll
    for (int ks = 0; ks < 3; ++ks)
      acc = wmma16(frag_a_rm(sX, XS, m0, ks * 32), frag_b_t(w1, DIM, ks * 32, n0), acc);
    int lane = tid & 31;
    int n = n0 + (lane & 15);
    int mb = m0 + ((lane >> 4) << 3);
    float bn = fc1_b[n];
#pragma unroll
    for (int r = 0; r < 8; ++r) {
      float v = acc[r] + bn;
      v = 0.5f * v * (1.f + erff(v * 0.70710678f));
      sH[(mb + r) * HS + n] = (_Float16)v;
    }
  }
  __syncthreads();

  // fc2: (64x384)@(384x96), 24 tiles, 3 per wave, K=12 steps
  for (int ti = wave * 3; ti < wave * 3 + 3; ++ti) {
    int m0 = (ti % 4) << 4;
    int n0 = (ti / 4) << 4;
    v8f acc = {};
#pragma unroll
    for (int ks = 0; ks < 12; ++ks)
      acc = wmma16(frag_a_rm(sH, HS, m0, ks * 32), frag_b_t(w2, 384, ks * 32, n0), acc);
    int lane = tid & 31;
    int n = n0 + (lane & 15);
    int mb = m0 + ((lane >> 4) << 3);
    float bn = fc2_b[n];
#pragma unroll
    for (int r = 0; r < 8; ++r) sO2[(mb + r) * OS2 + n] = acc[r] + bn;
  }
  __syncthreads();

  if (tid < NTOK) {
    float s = 0.f, s2 = 0.f;
    for (int c = 0; c < DIM; ++c) { float v = sO2[tid * OS2 + c]; s += v; s2 += v * v; }
    float mu = s / DIM;
    sMu[tid] = mu;
    sRstd[tid] = rsqrtf(s2 / DIM - mu * mu + 1e-5f);
  }
  __syncthreads();
  for (int i = tid; i < NTOK * DIM; i += 256) {
    int t = i / DIM, c = i % DIM;
    float v = (sO2[t * OS2 + c] - sMu[t]) * sRstd[t] * n2g[c] + n2b[c];
    out[(row0 + t) * DIM + c] += v;
  }
}

extern "C" void kernel_launch(void* const* d_in, const int* in_sizes, int n_in,
                              void* d_out, int out_size, void* d_ws, size_t ws_size,
                              hipStream_t stream) {
  (void)in_sizes; (void)n_in; (void)out_size; (void)ws_size;
  const float* x        = (const float*)d_in[0];
  const float* qkv_w    = (const float*)d_in[1];
  const float* q_bias   = (const float*)d_in[2];
  const float* v_bias   = (const float*)d_in[3];
  const float* lscale   = (const float*)d_in[4];
  const float* cpb_w1   = (const float*)d_in[5];
  const float* cpb_b1   = (const float*)d_in[6];
  const float* cpb_w2   = (const float*)d_in[7];
  const float* proj_w   = (const float*)d_in[8];
  const float* proj_b   = (const float*)d_in[9];
  const float* n1g      = (const float*)d_in[10];
  const float* n1b      = (const float*)d_in[11];
  const float* fc1_w    = (const float*)d_in[12];
  const float* fc1_b    = (const float*)d_in[13];
  const float* fc2_w    = (const float*)d_in[14];
  const float* fc2_b    = (const float*)d_in[15];
  const float* n2g      = (const float*)d_in[16];
  const float* n2b      = (const float*)d_in[17];
  float* out = (float*)d_out;

  _Float16* w16 = (_Float16*)d_ws;                                   // 110592 f16
  float* rpb  = (float*)((char*)d_ws + 221184);                      // 3*64*64 f32
  float* qkvb = (float*)((char*)d_ws + 221184 + 49152);              // 288 f32

  convert_w_kernel<<<432, 256, 0, stream>>>(qkv_w, proj_w, fc1_w, fc2_w,
                                            q_bias, v_bias, w16, qkvb);
  rpb_kernel<<<1, 256, 0, stream>>>(cpb_w1, cpb_b1, cpb_w2, rpb);
  swin_attn_kernel<<<NWIN, 256, 0, stream>>>(x, w16, qkvb, lscale,
                                             proj_b, n1g, n1b, rpb, out);
  swin_mlp_kernel<<<NWIN, 256, 0, stream>>>(w16, fc1_b, fc2_b, n2g, n2b, out);
}
